// MSVA_base_52948356825146
// MI455X (gfx1250) — compile-verified, matrix operand order checked by reading, not run on hardware
//
#include <hip/hip_runtime.h>
#include <hip/hip_bf16.h>

// ---------------------------------------------------------------------------
// Types for CDNA5 WMMA
// ---------------------------------------------------------------------------
typedef __attribute__((ext_vector_type(16))) __bf16 v16bf;
typedef __attribute__((ext_vector_type(8)))  float  v8f;

union FragBF {
    v16bf v;
    unsigned short u[16];
    uint4 q[2];
};

// float -> bf16 (round-to-nearest-even), as raw ushort bits
__device__ __forceinline__ unsigned short f2bf(float f) {
    union { float f; unsigned u; } cv; cv.f = f;
    unsigned u = cv.u;
    u += 0x7fffu + ((u >> 16) & 1u);
    return (unsigned short)(u >> 16);
}
__device__ __forceinline__ float bf2f(unsigned short h) {
    union { unsigned u; float f; } cv; cv.u = ((unsigned)h) << 16;
    return cv.f;
}

// Load a 16(rows)x32(K) bf16 fragment given a per-lane row pointer
// (caller supplies rowptr = base + rowIdx*ld + k0).
// ISA layout: lane half picks K sub-block; elements 0..7 = K g0 (+half*8),
// elements 8..15 = K 16+half*8 .. — two contiguous 16B loads per lane.
__device__ __forceinline__ v16bf load_frag_rowptr(const unsigned short* p) {
    int half = (threadIdx.x >> 4) & 1;
    const unsigned short* q = p + half * 8;
    FragBF f;
    f.q[0] = *(const uint4*)(q);
    f.q[1] = *(const uint4*)(q + 16);
    return f.v;
}

// Row-major A/B(NT) fragment: rows are lane&15 starting at `tile`
__device__ __forceinline__ v16bf load_frag_rm(const unsigned short* tile, int ld) {
    int row = threadIdx.x & 15;
    return load_frag_rowptr(tile + (size_t)row * ld);
}

__device__ __forceinline__ v8f wmma_bf16(v16bf a, v16bf b, v8f c) {
    return __builtin_amdgcn_wmma_f32_16x16x32_bf16(false, a, false, b, (short)0, c, false, false);
}

// ---------------------------------------------------------------------------
// fp32 -> bf16 conversion (grid-stride)
// ---------------------------------------------------------------------------
__global__ void __launch_bounds__(256)
msva_cvt_bf16(const float* __restrict__ in, unsigned short* __restrict__ out, long n) {
    long i = (long)blockIdx.x * blockDim.x + threadIdx.x;
    long stride = (long)gridDim.x * blockDim.x;
    for (; i < n; i += stride) out[i] = f2bf(in[i]);
}

// ---------------------------------------------------------------------------
// GEMM (NT): C[m,n] = act(alpha * sum_k A[m,k]*Bm[n,k] + bias[n])
// A: bf16 [M x K], Bm: bf16 [Nout x K], both K-contiguous. Wave tile: 16x64.
// ---------------------------------------------------------------------------
__global__ void __launch_bounds__(256)
msva_gemm_nt(const unsigned short* __restrict__ A, const unsigned short* __restrict__ Bm,
             const float* __restrict__ bias, float* __restrict__ outF,
             unsigned short* __restrict__ outH,
             int M, int Nout, int K, float alpha, int relu) {
    int wid = (int)((blockIdx.x * blockDim.x + threadIdx.x) >> 5);
    int tilesN = Nout >> 6;
    int tm = wid / tilesN, tn = wid % tilesN;
    if (tm * 16 >= M) return;
    int m0 = tm * 16, n0 = tn * 64;

    v8f acc[4] = {v8f{}, v8f{}, v8f{}, v8f{}};
    const unsigned short* arow = A + (size_t)m0 * K;
    const unsigned short* b0r = Bm + (size_t)(n0 +  0) * K;
    const unsigned short* b1r = Bm + (size_t)(n0 + 16) * K;
    const unsigned short* b2r = Bm + (size_t)(n0 + 32) * K;
    const unsigned short* b3r = Bm + (size_t)(n0 + 48) * K;

    for (int k0 = 0; k0 < K; k0 += 32) {
        __builtin_prefetch(arow + k0 + 64, 0, 1);
        v16bf a  = load_frag_rm(arow + k0, K);
        v16bf b0 = load_frag_rm(b0r + k0, K);
        v16bf b1 = load_frag_rm(b1r + k0, K);
        v16bf b2 = load_frag_rm(b2r + k0, K);
        v16bf b3 = load_frag_rm(b3r + k0, K);
        acc[0] = wmma_bf16(a, b0, acc[0]);
        acc[1] = wmma_bf16(a, b1, acc[1]);
        acc[2] = wmma_bf16(a, b2, acc[2]);
        acc[3] = wmma_bf16(a, b3, acc[3]);
    }

    int lane = threadIdx.x & 31, col = lane & 15, half = lane >> 4;
#pragma unroll
    for (int c = 0; c < 4; ++c) {
#pragma unroll
        for (int e = 0; e < 8; ++e) {
            int m = m0 + half * 8 + e;
            int n = n0 + c * 16 + col;
            float v = alpha * acc[c][e];
            if (bias) v += bias[n];
            if (relu) v = fmaxf(v, 0.f);
            size_t idx = (size_t)m * Nout + n;
            if (outF) outF[idx] = v;
            if (outH) outH[idx] = f2bf(v);
        }
    }
}

// ---------------------------------------------------------------------------
// Banded logits: S[b][m][n-m+255] = Q[m] . K[n] for |m-n| < 256
// Wave tile: 16(m) x 16(n), K = D loop via WMMA.
// ---------------------------------------------------------------------------
__global__ void __launch_bounds__(256)
msva_band_logits(const unsigned short* __restrict__ Qb, const unsigned short* __restrict__ Kb,
                 float* __restrict__ S, int Bn, int Nn, int Dd) {
    const int TBLK = 34;
    int wid = (int)((blockIdx.x * blockDim.x + threadIdx.x) >> 5);
    int mblocks = Nn >> 4;
    int per_b = mblocks * TBLK;
    int b = wid / per_b;
    if (b >= Bn) return;
    int r = wid % per_b;
    int m0 = (r / TBLK) * 16;
    int t  = r % TBLK;
    int n0 = m0 - 256 + t * 16;
    if (n0 + 15 < 0 || n0 >= Nn) return;
    if (n0 + 15 - m0 <= -256 || n0 - (m0 + 15) >= 256) return;  // fully masked

    int lane = threadIdx.x & 31, col = lane & 15, half = lane >> 4;
    const unsigned short* qrow = Qb + (size_t)(b * Nn + m0 + col) * Dd;     // A row per lane
    int nrow = n0 + col;
    int ncl  = nrow < 0 ? 0 : (nrow >= Nn ? Nn - 1 : nrow);                 // clamp (masked later)
    const unsigned short* krow = Kb + (size_t)(b * Nn + ncl) * Dd;

    v8f acc = {};
    for (int k0 = 0; k0 < Dd; k0 += 32) {
        v16bf a = load_frag_rowptr(qrow + k0);
        v16bf bk = load_frag_rowptr(krow + k0);
        acc = wmma_bf16(a, bk, acc);
    }
#pragma unroll
    for (int e = 0; e < 8; ++e) {
        int m = m0 + half * 8 + e;
        int n = n0 + col;
        int d = n - m;
        if (n >= 0 && n < Nn && d > -256 && d < 256)
            S[((size_t)(b * Nn + m)) * 512 + (d + 255)] = acc[e];
    }
}

// ---------------------------------------------------------------------------
// Row softmax over the band; writes normalized probs (bf16) for all 512 j,
// zeros outside the valid band. One wave per row.
// ---------------------------------------------------------------------------
__global__ void __launch_bounds__(256)
msva_band_softmax(const float* __restrict__ S, unsigned short* __restrict__ P, int Bn, int Nn) {
    int row = (int)((blockIdx.x * blockDim.x + threadIdx.x) >> 5);
    if (row >= Bn * Nn) return;
    int m = row % Nn;
    int lane = threadIdx.x & 31;
    int jlo = 255 - m;          if (jlo < 0) jlo = 0;
    int jhi = (Nn - 1 - m) + 255; if (jhi > 510) jhi = 510;
    const float* s = S + (size_t)row * 512;

    float vals[16];
    float mx = -3.0e38f;
#pragma unroll
    for (int i = 0; i < 16; ++i) {
        int j = lane + i * 32;
        float v = (j >= jlo && j <= jhi) ? s[j] : -3.0e38f;
        vals[i] = v;
        mx = fmaxf(mx, v);
    }
    for (int off = 16; off > 0; off >>= 1) mx = fmaxf(mx, __shfl_xor(mx, off, 32));

    float sum = 0.f;
#pragma unroll
    for (int i = 0; i < 16; ++i) {
        int j = lane + i * 32;
        float e = (j >= jlo && j <= jhi) ? __expf(vals[i] - mx) : 0.f;
        vals[i] = e;
        sum += e;
    }
    for (int off = 16; off > 0; off >>= 1) sum += __shfl_xor(sum, off, 32);
    float inv = 1.f / sum;
#pragma unroll
    for (int i = 0; i < 16; ++i) {
        int j = lane + i * 32;
        P[(size_t)row * 512 + j] = f2bf(vals[i] * inv);
    }
}

// ---------------------------------------------------------------------------
// Transposed-weight combine: Y[b,n,d] = sum_m P[b][m][n-m+255] * V[b][m][d]
// Wave tile: 16(n) x 64(d); m swept in 32-chunks via WMMA.
// ---------------------------------------------------------------------------
__global__ void __launch_bounds__(256)
msva_band_attn_out(const unsigned short* __restrict__ P, const unsigned short* __restrict__ Vb,
                   unsigned short* __restrict__ Y, int Bn, int Nn, int Dd) {
    int wid = (int)((blockIdx.x * blockDim.x + threadIdx.x) >> 5);
    int tilesD = Dd >> 6;
    int per_b = (Nn >> 4) * tilesD;
    int b = wid / per_b;
    if (b >= Bn) return;
    int r = wid % per_b;
    int n0 = (r / tilesD) * 16;
    int d0 = (r % tilesD) * 64;
    int lane = threadIdx.x & 31, rloc = lane & 15, half = lane >> 4;

    v8f acc[4] = {v8f{}, v8f{}, v8f{}, v8f{}};

    for (int it = 0; it < 17; ++it) {
        int mbase = n0 - 256 + it * 32;
        // A fragment: A[r][kk] = P[m = mbase+kk][ (n0+r) - m + 255 ]
        FragBF fa;
#pragma unroll
        for (int e = 0; e < 16; ++e) {
            int g = e >> 3;
            int kk = g * 16 + half * 8 + (e & 7);
            int m = mbase + kk;
            int j = (n0 + rloc) - m + 255;
            unsigned short u = 0;
            if (m >= 0 && m < Nn && j >= 0 && j < 512)
                u = P[((size_t)(b * Nn + m)) * 512 + j];
            fa.u[e] = u;
        }
        // B fragments (NN layout): B[k][n] = V[mbase+k][d0 + c*16 + rloc]
#pragma unroll
        for (int c = 0; c < 4; ++c) {
            FragBF fb;
#pragma unroll
            for (int e = 0; e < 16; ++e) {
                int g = e >> 3;
                int k = g * 16 + half * 8 + (e & 7);
                int m = mbase + k;
                int mc = m < 0 ? 0 : (m >= Nn ? Nn - 1 : m);    // A==0 masks OOB rows
                fb.u[e] = Vb[(size_t)(b * Nn + mc) * Dd + d0 + c * 16 + rloc];
            }
            acc[c] = wmma_bf16(fa.v, fb.v, acc[c]);
        }
    }
#pragma unroll
    for (int c = 0; c < 4; ++c) {
#pragma unroll
        for (int e = 0; e < 8; ++e) {
            int n = n0 + half * 8 + e;
            int d = d0 + c * 16 + rloc;
            Y[(size_t)(b * Nn + n) * Dd + d] = f2bf(acc[c][e]);
        }
    }
}

// ---------------------------------------------------------------------------
// Row LayerNorm (ddof=1, eps on std). Optional residual add, optional fp32
// accumulation (init or +=), optional bf16 output. One 256-thread block/row.
// ---------------------------------------------------------------------------
__device__ __forceinline__ float msva_block_sum(float v) {
    __shared__ float red[8];
    for (int off = 16; off > 0; off >>= 1) v += __shfl_xor(v, off, 32);
    int warp = threadIdx.x >> 5, lane = threadIdx.x & 31;
    if (lane == 0) red[warp] = v;
    __syncthreads();
    float s = 0.f;
#pragma unroll
    for (int i = 0; i < 8; ++i) s += red[i];
    __syncthreads();
    return s;
}

__global__ void __launch_bounds__(256)
msva_ln_rows(const float* __restrict__ in, const float* __restrict__ resid,
             const float* __restrict__ g, const float* __restrict__ bb,
             float* __restrict__ accum, unsigned short* __restrict__ outH,
             int Dd, int initAcc) {
    int row = blockIdx.x;
    const float* x = in + (size_t)row * Dd;
    const float* rr = resid ? resid + (size_t)row * Dd : nullptr;
    float t[4];
    float s = 0.f;
#pragma unroll
    for (int i = 0; i < 4; ++i) {
        int idx = threadIdx.x + i * 256;
        float v = x[idx] + (rr ? rr[idx] : 0.f);
        t[i] = v; s += v;
    }
    float total = msva_block_sum(s);
    float mean = total / (float)Dd;
    float s2 = 0.f;
#pragma unroll
    for (int i = 0; i < 4; ++i) { float d = t[i] - mean; s2 += d * d; }
    float var = msva_block_sum(s2) / (float)(Dd - 1);
    float denom = sqrtf(var) + 1e-6f;
#pragma unroll
    for (int i = 0; i < 4; ++i) {
        int idx = threadIdx.x + i * 256;
        float o = g[idx] * (t[i] - mean) / denom + bb[idx];
        size_t oi = (size_t)row * Dd + idx;
        if (accum) accum[oi] = initAcc ? o : (accum[oi] + o);
        if (outH) outH[oi] = f2bf(o);
    }
}

// ---------------------------------------------------------------------------
// Final projection to OUT=1: out[row] = sum_k H[row,k]*w[k] + b. Wave per row.
// ---------------------------------------------------------------------------
__global__ void __launch_bounds__(256)
msva_final_dot(const unsigned short* __restrict__ H, const float* __restrict__ w,
               const float* __restrict__ bptr, float* __restrict__ out, int rows, int Dd) {
    int row = (int)((blockIdx.x * blockDim.x + threadIdx.x) >> 5);
    if (row >= rows) return;
    int lane = threadIdx.x & 31;
    float s = 0.f;
    for (int i = lane; i < Dd; i += 32)
        s += bf2f(H[(size_t)row * Dd + i]) * w[i];
    for (int off = 16; off > 0; off >>= 1) s += __shfl_xor(s, off, 32);
    if (lane == 0) out[row] = s + bptr[0];
}

// ---------------------------------------------------------------------------
// Host-side orchestration
// ---------------------------------------------------------------------------
extern "C" void kernel_launch(void* const* d_in, const int* in_sizes, int n_in,
                              void* d_out, int out_size, void* d_ws, size_t ws_size,
                              hipStream_t stream) {
    constexpr int B = 2, N = 4096, D = 1024, M = B * N;   // M = 8192
    constexpr float Q_SCALE = 0.06f;
    constexpr long WELEM = (long)D * D;

    const float* x[3]  = {(const float*)d_in[0], (const float*)d_in[1], (const float*)d_in[2]};
    const float* Wsq[7] = {(const float*)d_in[3],  // Wk
                           (const float*)d_in[4],  // Wq
                           (const float*)d_in[5],  // Wv
                           (const float*)d_in[6],  // Wo
                           (const float*)d_in[7],  // ka_w
                           (const float*)d_in[9],  // kb_w
                           (const float*)d_in[11]};// kc_w
    const float* ka_b = (const float*)d_in[8];
    const float* kb_b = (const float*)d_in[10];
    const float* kc_b = (const float*)d_in[12];
    const float* kd_w = (const float*)d_in[13];
    const float* kd_b = (const float*)d_in[14];
    const float* g1   = (const float*)d_in[15];
    const float* b1   = (const float*)d_in[16];
    const float* g2   = (const float*)d_in[17];
    const float* b2   = (const float*)d_in[18];

    char* ws = (char*)d_ws;
    const size_t MB = 1024ull * 1024ull;
    unsigned short* wbf[7];
    for (int i = 0; i < 7; ++i) wbf[i] = (unsigned short*)(ws + (size_t)i * 2 * MB);
    unsigned short* xbf   = (unsigned short*)(ws + 16 * MB);    // 16 MB bf16 [M x D]
    unsigned short* Kbf   = (unsigned short*)(ws + 32 * MB);    // 16 MB
    unsigned short* Qbf   = (unsigned short*)(ws + 48 * MB);    // 16 MB
    unsigned short* Vbf   = (unsigned short*)(ws + 64 * MB);    // 16 MB
    float*          Sbuf  = (float*)         (ws + 80 * MB);    // 16 MB [M x 512] f32
    unsigned short* Pbuf  = (unsigned short*)(ws + 96 * MB);    //  8 MB [M x 512] bf16
    unsigned short* Attn  = (unsigned short*)(ws + 104 * MB);   // 16 MB bf16
    float*          Af32  = (float*)         (ws + 120 * MB);   // 32 MB f32 [M x D]
    float*          Ysum  = (float*)         (ws + 152 * MB);   // 32 MB f32 [M x D]

    const long NXD = (long)M * D;
    const int gemmBlocks = ((M / 16) * (D / 64)) / 8;             // 1024 blocks of 8 waves
    const int logitBlocks = (B * (N / 16) * 34 + 7) / 8;          // 16x16 band tiles
    const int rowWaveBlocks = (M + 7) / 8;                        // wave-per-row kernels

    // Weights -> bf16 (once per launch)
    for (int i = 0; i < 7; ++i)
        msva_cvt_bf16<<<1024, 256, 0, stream>>>(Wsq[i], wbf[i], WELEM);

    for (int ix = 0; ix < 3; ++ix) {
        msva_cvt_bf16<<<2048, 256, 0, stream>>>(x[ix], xbf, NXD);
        // K, Q (scaled), V projections
        msva_gemm_nt<<<gemmBlocks, 256, 0, stream>>>(xbf, wbf[0], nullptr, nullptr, Kbf, M, D, D, 1.0f, 0);
        msva_gemm_nt<<<gemmBlocks, 256, 0, stream>>>(xbf, wbf[1], nullptr, nullptr, Qbf, M, D, D, Q_SCALE, 0);
        msva_gemm_nt<<<gemmBlocks, 256, 0, stream>>>(xbf, wbf[2], nullptr, nullptr, Vbf, M, D, D, 1.0f, 0);
        // Banded attention
        msva_band_logits<<<logitBlocks, 256, 0, stream>>>(Qbf, Kbf, Sbuf, B, N, D);
        msva_band_softmax<<<rowWaveBlocks, 256, 0, stream>>>(Sbuf, Pbuf, B, N);
        msva_band_attn_out<<<gemmBlocks, 256, 0, stream>>>(Pbuf, Vbf, Attn, B, N, D);
        // Output projection, residual + LN, accumulate
        msva_gemm_nt<<<gemmBlocks, 256, 0, stream>>>(Attn, wbf[3], nullptr, Af32, nullptr, M, D, D, 1.0f, 0);
        msva_ln_rows<<<M, 256, 0, stream>>>(Af32, x[ix], g1, b1, Ysum, nullptr, D, ix == 0 ? 1 : 0);
    }

    // MLP head
    msva_cvt_bf16<<<2048, 256, 0, stream>>>(Ysum, xbf, NXD);
    msva_gemm_nt<<<gemmBlocks, 256, 0, stream>>>(xbf, wbf[4], ka_b, nullptr, Kbf, M, D, D, 1.0f, 0);
    msva_gemm_nt<<<gemmBlocks, 256, 0, stream>>>(Kbf, wbf[5], kb_b, nullptr, Vbf, M, D, D, 1.0f, 0);
    msva_gemm_nt<<<gemmBlocks, 256, 0, stream>>>(Vbf, wbf[6], kc_b, Af32, nullptr, M, D, D, 1.0f, 1);
    msva_ln_rows<<<M, 256, 0, stream>>>(Af32, nullptr, g2, b2, nullptr, Qbf, D, 0);
    msva_final_dot<<<rowWaveBlocks, 256, 0, stream>>>(Qbf, kd_w, kd_b, (float*)d_out, M, D);
}
